// FasterRCNN_14937896256268
// MI455X (gfx1250) — matrix operand order repaired and checked by reading, not compiled, not run on hardware
//
#include <hip/hip_runtime.h>
#include <math.h>

// ---------------------------------------------------------------------------
// Faster-RCNN head for MI455X (gfx1250, wave32, WMMA).
// FC1 dominates (205 GFLOP / 411 MB weights) -> bf16 WMMA with on-the-fly
// f32->bf16 weight conversion during LDS staging (single HBM pass of W).
// A tile is staged with GLOBAL_LOAD_ASYNC_TO_LDS (ASYNCcnt) when available.
// Workspace layout (needs ~76.6 MB):
//   X  : bf16 [1024 x 25088]   (pooled features, rows 1000..1023 zero)
//   H1 : bf16 [1024 x 4096]
//   H2 : f32  [1024 x 4096]
// ---------------------------------------------------------------------------

typedef __bf16 bf16_t;
typedef __attribute__((ext_vector_type(16))) __bf16 v16bf;
typedef __attribute__((ext_vector_type(8)))  __bf16 v8bf;
typedef __attribute__((ext_vector_type(8)))  float  v8f;
typedef int v4i_ __attribute__((vector_size(4 * sizeof(int))));

#define NROI      1000
#define MPAD      1024
#define FH        38
#define FW        50
#define FC        512
#define KFEAT     25088          // 7*7*512
#define NH        4096

#if defined(__gfx1250__) && \
    __has_builtin(__builtin_amdgcn_global_load_async_to_lds_b128) && \
    __has_builtin(__builtin_amdgcn_s_wait_asynccnt)
#define USE_ASYNC_LDS 1
// async param types per hipcc diagnostic: (v4i AS1*, v4i AS3*, imm, imm)
typedef __attribute__((address_space(1))) v4i_* gptr_t;
typedef __attribute__((address_space(3))) v4i_* lptr_t;
#else
#define USE_ASYNC_LDS 0
#endif

__device__ __forceinline__ int iclamp(int v, int lo, int hi) {
    return v < lo ? lo : (v > hi ? hi : v);
}

// ---------------------------------------------------------------------------
// Kernel 1: fused crop_and_resize(14x14) + 2x2 maxpool -> bf16 X row
// One block per ROI row (incl. zero-padding rows), 256 threads = 2 channels ea.
// ---------------------------------------------------------------------------
__global__ __launch_bounds__(256)
void crop_pool_kernel(const float* __restrict__ feats,
                      const float* __restrict__ boxes,
                      bf16_t* __restrict__ X)
{
    const int n = blockIdx.x;
    const int t = threadIdx.x;
    bf16_t* xr = X + (size_t)n * KFEAT;

    if (n >= NROI) {                       // zero the M-padding rows
        for (int i = t; i < KFEAT; i += 256) xr[i] = (bf16_t)0.0f;
        return;
    }

    const float y1 = boxes[n * 4 + 0];
    const float x1 = boxes[n * 4 + 1];
    const float y2 = boxes[n * 4 + 2];
    const float x2 = boxes[n * 4 + 3];
    const float by = y1 * (float)(FH - 1);
    const float bx = x1 * (float)(FW - 1);
    const float sy = (y2 - y1) * (float)(FH - 1) / 13.0f;
    const float sx = (x2 - x1) * (float)(FW - 1) / 13.0f;

    const int c0 = t * 2;                  // two channels per thread

    for (int cell = 0; cell < 49; ++cell) {
        const int py = cell / 7, px = cell % 7;
        float m0 = -INFINITY, m1 = -INFINITY;
        #pragma unroll
        for (int s = 0; s < 4; ++s) {
            const int ii = 2 * py + (s >> 1);
            const int jj = 2 * px + (s & 1);
            const float ys = by + (float)ii * sy;
            const float xs = bx + (float)jj * sx;
            const float fy = floorf(ys), fx = floorf(xs);
            const float wy = ys - fy,    wx = xs - fx;
            const int y0 = iclamp((int)fy, 0, FH - 1);
            const int y1i = iclamp(y0 + 1, 0, FH - 1);
            const int x0 = iclamp((int)fx, 0, FW - 1);
            const int x1i = iclamp(x0 + 1, 0, FW - 1);
            const float w00 = (1.f - wy) * (1.f - wx);
            const float w01 = (1.f - wy) * wx;
            const float w10 = wy * (1.f - wx);
            const float w11 = wy * wx;
            const float* f00 = feats + ((y0  * FW + x0 ) * FC + c0);
            const float* f01 = feats + ((y0  * FW + x1i) * FC + c0);
            const float* f10 = feats + ((y1i * FW + x0 ) * FC + c0);
            const float* f11 = feats + ((y1i * FW + x1i) * FC + c0);
            const float v0 = f00[0]*w00 + f01[0]*w01 + f10[0]*w10 + f11[0]*w11;
            const float v1 = f00[1]*w00 + f01[1]*w01 + f10[1]*w10 + f11[1]*w11;
            m0 = fmaxf(m0, v0);
            m1 = fmaxf(m1, v1);
        }
        xr[cell * FC + c0]     = (bf16_t)m0;
        xr[cell * FC + c0 + 1] = (bf16_t)m1;
    }
}

// ---------------------------------------------------------------------------
// Kernel 2: GEMM  out[M,N] = act(A[M,K](bf16) @ W[K,N](f32->bf16) + bias)
// Block tile 128x128, BK=32, 8 waves (wave32) each owning a 64x32 wave tile
// = 4x2 fragments of v_wmma_f32_16x16x32_bf16 per k-step.
//  - K, N are compile-time so stride offsets fold into load IOFFSETs.
//  - A tile: async global->LDS copy (bf16 verbatim) when builtins exist.
//  - B tile: f32 W, coalesced stride-N scalar loads, cvt to bf16 in-register,
//    stored TRANSPOSED in LDS (Bs[col][k]) so B fragments are ds_load_b128.
// ---------------------------------------------------------------------------
template<bool RELU, bool OUTBF, int K, int N>
__global__ __launch_bounds__(256)
void gemm_bf16_kernel(const bf16_t* __restrict__ A,
                      const float*  __restrict__ W,
                      const float*  __restrict__ bias,
                      void* __restrict__ out)
{
    __shared__ __align__(16) bf16_t As[128][40];    // [row][k]  +8 pad (80B)
    __shared__ __align__(16) bf16_t Bs[128][40];    // [col][k]  +8 pad (80B)

    const int t    = threadIdx.x;
    const int lane = t & 31;
    const int wave = t >> 5;          // 0..7
    const int wm   = wave >> 2;       // 0..1 -> 64-row strip
    const int wn   = wave & 3;        // 0..3 -> 32-col strip

    const int mrow0 = blockIdx.x * 128;
    const int ncol0 = blockIdx.y * 128;

    v8f acc[4][2];
    #pragma unroll
    for (int i = 0; i < 4; ++i)
        #pragma unroll
        for (int j = 0; j < 2; ++j)
            acc[i][j] = (v8f){0.f, 0.f, 0.f, 0.f, 0.f, 0.f, 0.f, 0.f};

    const int ar  = t >> 1;           // A stage: row 0..127
    const int ak  = (t & 1) << 4;     //          k-chunk 0 / 16
    const int bcl = t & 127;          // B stage: column 0..127
    const int bkh = (t >> 7) << 4;    //          k-half 0 / 16

    // hoisted base pointers, advanced by constant strides each k-step
    const bf16_t* agp = A + (size_t)(mrow0 + ar) * K + ak;
    const float*  bgp = W + (size_t)bkh * N + ncol0 + bcl;
    bf16_t* aldst = &As[ar][ak];

    for (int k0 = 0; k0 < K; k0 += 32) {
        // ---- stage A tile (bf16 verbatim, 32B per thread, async->LDS) ----
#if USE_ASYNC_LDS
        __builtin_amdgcn_global_load_async_to_lds_b128(
            (gptr_t)agp, (lptr_t)aldst, 0, 0);
        __builtin_amdgcn_global_load_async_to_lds_b128(
            (gptr_t)agp, (lptr_t)aldst, 16, 0);
#else
        {
            const uint4* gp = (const uint4*)agp;
            uint4 u0 = gp[0];
            uint4 u1 = gp[1];
            *(uint4*)aldst       = u0;
            *(uint4*)(aldst + 8) = u1;
        }
#endif
        // ---- stage B tile transposed (f32 -> bf16 in flight) ----
        // thread owns (col=bcl, k=bkh..bkh+15); global reads coalesce across
        // the 128 adjacent lanes with constant IOFFSETs; LDS store is 2x16B.
        {
            v8bf lo, hi;
            #pragma unroll
            for (int e = 0; e < 8; ++e) lo[e] = (bf16_t)bgp[e * N];
            #pragma unroll
            for (int e = 0; e < 8; ++e) hi[e] = (bf16_t)bgp[(8 + e) * N];
            *(v8bf*)&Bs[bcl][bkh]     = lo;
            *(v8bf*)&Bs[bcl][bkh + 8] = hi;
        }
        // prefetch next W tile into the cache hierarchy (global_prefetch_b8)
        if (k0 + 32 < K)
            __builtin_prefetch(bgp + 32 * N, 0, 0);

#if USE_ASYNC_LDS
        __builtin_amdgcn_s_wait_asynccnt(0);
#endif
        __syncthreads();

        // ---- fragment loads (ISA 7.12.2 16-bit A layout: lanes0-15 hold
        //      K 0-7 & 16-23, lanes16-31 hold K 8-15 & 24-31) ----
        const int abase = (lane < 16) ? 0 : 8;
        const int arow0 = wm * 64 + (lane & 15);
        v16bf afr[4];
        #pragma unroll
        for (int i = 0; i < 4; ++i) {
            const int row = arow0 + i * 16;
            v8bf lo = *(const v8bf*)&As[row][abase];
            v8bf hi = *(const v8bf*)&As[row][abase + 16];
            v16bf a;
            #pragma unroll
            for (int e = 0; e < 8; ++e) { a[e] = lo[e]; a[8 + e] = hi[e]; }
            afr[i] = a;
        }
        // B layout: lanes0-15 -> K 0-7 & 16-23, lanes16-31 -> K 8-15 & 24-31, col = lane&15.
        // Bs is [col][k] so each lane's two 8-K chunks are contiguous.
        const int kb   = (lane < 16) ? 0 : 8;
        const int bcol = wn * 32 + (lane & 15);
        v16bf bfr[2];
        #pragma unroll
        for (int j = 0; j < 2; ++j) {
            const int col = bcol + j * 16;
            v8bf lo = *(const v8bf*)&Bs[col][kb];
            v8bf hi = *(const v8bf*)&Bs[col][kb + 16];
            v16bf b;
            #pragma unroll
            for (int e = 0; e < 8; ++e) { b[e] = lo[e]; b[8 + e] = hi[e]; }
            bfr[j] = b;
        }

        #pragma unroll
        for (int i = 0; i < 4; ++i)
            #pragma unroll
            for (int j = 0; j < 2; ++j)
                acc[i][j] = __builtin_amdgcn_wmma_f32_16x16x32_bf16(
                    false, afr[i], false, bfr[j], (short)0, acc[i][j],
                    false, false);

        __syncthreads();

        agp += 32;                    // A advances 32 k-elements
        bgp += 32 * N;                // W advances 32 rows
    }

    // ---- epilogue: D layout vgpr r -> row (lane/16)*8 + r, col lane&15 ----
    const int rr   = (lane >> 4) << 3;
    const int ccol = lane & 15;
    #pragma unroll
    for (int j = 0; j < 2; ++j) {
        const int gcol = ncol0 + wn * 32 + j * 16 + ccol;
        const float bj = bias[gcol];
        #pragma unroll
        for (int i = 0; i < 4; ++i) {
            #pragma unroll
            for (int r = 0; r < 8; ++r) {
                const int grow = mrow0 + wm * 64 + i * 16 + rr + r;
                float v = acc[i][j][r] + bj;
                if (RELU) v = fmaxf(v, 0.0f);
                if constexpr (OUTBF) {
                    ((bf16_t*)out)[(size_t)grow * N + gcol] = (bf16_t)v;
                } else {
                    ((float*)out)[(size_t)grow * N + gcol] = v;
                }
            }
        }
    }
}

// ---------------------------------------------------------------------------
// Kernel 3: cls/reg heads + softmax. One block per ROI, x row in LDS,
// threads j<21 -> cls col j (stride-21 reads coalesce across lanes),
// threads 21..100 -> reg col j-21.
// ---------------------------------------------------------------------------
__global__ __launch_bounds__(128)
void heads_kernel(const float* __restrict__ H2,
                  const float* __restrict__ Wc, const float* __restrict__ bc,
                  const float* __restrict__ Wr, const float* __restrict__ br,
                  float* __restrict__ out)
{
    __shared__ float xs[NH];
    __shared__ float lg[101];
    __shared__ float red[2];

    const int n = blockIdx.x;
    const int t = threadIdx.x;
    const float* xrow = H2 + (size_t)n * NH;

    for (int k = t * 4; k < NH; k += 128 * 4)
        *(float4*)&xs[k] = *(const float4*)&xrow[k];
    __syncthreads();

    if (t < 101) {
        float acc;
        if (t < 21) {
            acc = bc[t];
            for (int k = 0; k < NH; ++k) acc += xs[k] * Wc[k * 21 + t];
        } else {
            const int j = t - 21;
            acc = br[j];
            for (int k = 0; k < NH; ++k) acc += xs[k] * Wr[k * 80 + j];
        }
        lg[t] = acc;
    }
    __syncthreads();

    if (t == 0) {
        float m = lg[0];
        for (int j = 1; j < 21; ++j) m = fmaxf(m, lg[j]);
        float s = 0.f;
        for (int j = 0; j < 21; ++j) s += expf(lg[j] - m);
        red[0] = m;
        red[1] = s;
    }
    __syncthreads();

    if (t < 21) {
        out[(size_t)n * 21 + t] = expf(lg[t] - red[0]) / red[1];
    } else if (t < 101) {
        out[(size_t)NROI * 21 + (size_t)n * 80 + (t - 21)] = lg[t];
    }
}

// ---------------------------------------------------------------------------
extern "C" void kernel_launch(void* const* d_in, const int* in_sizes, int n_in,
                              void* d_out, int out_size, void* d_ws, size_t ws_size,
                              hipStream_t stream)
{
    const float* feats = (const float*)d_in[0];
    const float* props = (const float*)d_in[1];
    const float* W1    = (const float*)d_in[2];
    const float* b1    = (const float*)d_in[3];
    const float* W2    = (const float*)d_in[4];
    const float* b2    = (const float*)d_in[5];
    const float* Wc    = (const float*)d_in[6];
    const float* bc    = (const float*)d_in[7];
    const float* Wr    = (const float*)d_in[8];
    const float* br    = (const float*)d_in[9];
    float* out = (float*)d_out;

    char* ws = (char*)d_ws;
    bf16_t* X  = (bf16_t*)ws;                                           // 51.4 MB
    bf16_t* H1 = (bf16_t*)(ws + (size_t)MPAD * KFEAT * sizeof(bf16_t)); //  8.4 MB
    float*  H2 = (float*)(ws + (size_t)MPAD * KFEAT * sizeof(bf16_t)
                             + (size_t)MPAD * NH * sizeof(bf16_t));     // 16.8 MB
    (void)in_sizes; (void)n_in; (void)out_size; (void)ws_size;

    // 1) crop + resize + maxpool -> bf16 X [1024 x 25088]
    crop_pool_kernel<<<MPAD, 256, 0, stream>>>(feats, props, X);

    // 2) H1 = relu(X @ W1 + b1)  (bf16 WMMA, 205.5 GFLOP)
    gemm_bf16_kernel<true, true, KFEAT, NH>
        <<<dim3(MPAD / 128, NH / 128), 256, 0, stream>>>(X, W1, b1, (void*)H1);

    // 3) H2 = relu(H1 @ W2 + b2) (bf16 WMMA, 33.6 GFLOP, f32 out)
    gemm_bf16_kernel<true, false, NH, NH>
        <<<dim3(MPAD / 128, NH / 128), 256, 0, stream>>>(H1, W2, b2, (void*)H2);

    // 4) cls softmax + reg heads -> d_out (cls[1000*21] ++ reg[1000*80])
    heads_kernel<<<NROI, 128, 0, stream>>>(H2, Wc, bc, Wr, br, out);
}
